// Wav2Vec2Quantizer_86681029968651
// MI455X (gfx1250) — matrix operand order repaired and checked by reading, not compiled, not run on hardware
//
#include <hip/hip_runtime.h>

// ---------------------------------------------------------------------------
// Problem constants (from the reference)
// ---------------------------------------------------------------------------
#define NTOK   32768          // B*T
#define HID    512
#define DOUT   256
#define NGRP   2
#define GD     128
#define NCV    320            // codevectors per group
#define VT     20             // NCV/16
#define KS2    4              // GD/32 k-steps in stage 2

// output layout (f32, concatenated in return order)
#define QF_BASE   ((size_t)0)                         // 32768*256
#define ENC_BASE  ((size_t)8388608)                   // + G*N*V
#define DIST_BASE ((size_t)29360128)                  // + G*N*V
#define PERP_IDX  ((size_t)50331648)

// workspace layout (bytes)
#define WS_HBF    ((size_t)0)                          // N*256 bf16 = 16 MB
#define WS_X2     (WS_HBF + (size_t)NTOK*DOUT*2)       // N*2 f32
#define WS_WPACK  (WS_X2 + (size_t)NTOK*NGRP*4)        // 512*256 bf16
#define WS_CPACK  (WS_WPACK + (size_t)HID*DOUT*2)      // 2*320*128 bf16
#define WS_C2     (WS_CPACK + (size_t)NGRP*NCV*GD*2)   // 640 f32
#define WS_CNT    (WS_C2 + (size_t)NGRP*NCV*4)         // 640 u32

typedef __attribute__((ext_vector_type(16))) __bf16 v16bf;
typedef __attribute__((ext_vector_type(8)))  __bf16 v8bf;
typedef __attribute__((ext_vector_type(8)))  float  v8f;

// Native f32 -> bf16 truncation: gfx1250 has v_cvt_pk_bf16_f32, let the
// backend select it instead of a manual shift/round VALU chain.
__device__ __forceinline__ __bf16 f2bf(float f) { return (__bf16)f; }

// ---------------------------------------------------------------------------
// Pre-pack W into the WMMA B-operand lane layout (bf16).
// Consumer index: ((kb*16 + nt)*32 + lane)*16 + t  -> 16 contiguous K per lane.
// ---------------------------------------------------------------------------
__global__ void prepack_w_kernel(const float* __restrict__ W, __bf16* __restrict__ wp) {
    int idx = blockIdx.x * blockDim.x + threadIdx.x;      // 0 .. 8191
    if (idx >= 16 * 16 * 32) return;
    int lane = idx & 31;
    int nt   = (idx >> 5) & 15;
    int kb   = idx >> 9;
    int half = lane >> 4;
    int n    = nt * 16 + (lane & 15);
    __bf16* dst = wp + (size_t)idx * 16;
    #pragma unroll
    for (int t = 0; t < 16; ++t) {
        int k = kb * 32 + 16 * half + t;
        dst[t] = f2bf(W[(size_t)k * DOUT + n]);
    }
}

// ---------------------------------------------------------------------------
// Pre-pack codevectors into B-operand layout (bf16).
// Consumer index: (((g*20 + vt)*4 + ks)*32 + lane)*16 + t
// ---------------------------------------------------------------------------
__global__ void prepack_c_kernel(const float* __restrict__ cv, __bf16* __restrict__ cp) {
    int idx = blockIdx.x * blockDim.x + threadIdx.x;      // 0 .. 5119
    if (idx >= NGRP * VT * KS2 * 32) return;
    int g    = idx / (VT * KS2 * 32);
    int rem  = idx % (VT * KS2 * 32);
    int vt   = rem / (KS2 * 32);
    int rem2 = rem % (KS2 * 32);
    int ks   = rem2 >> 5;
    int lane = rem2 & 31;
    int half = lane >> 4;
    int v    = vt * 16 + (lane & 15);
    const float* src = cv + ((size_t)g * NCV + v) * GD;
    __bf16* dst = cp + (size_t)idx * 16;
    #pragma unroll
    for (int t = 0; t < 16; ++t) {
        int k = ks * 32 + 16 * half + t;
        dst[t] = f2bf(src[k]);
    }
}

// ---------------------------------------------------------------------------
// c2[g][v] = sum_d codevectors^2 ; zero the usage counters (ws is poisoned).
// ---------------------------------------------------------------------------
__global__ void c2_counts_kernel(const float* __restrict__ cv,
                                 float* __restrict__ c2, unsigned* __restrict__ counts) {
    int gv = blockIdx.x * blockDim.x + threadIdx.x;
    if (gv >= NGRP * NCV) return;
    const float* row = cv + (size_t)gv * GD;
    float s = 0.f;
    #pragma unroll 8
    for (int d = 0; d < GD; ++d) s += row[d] * row[d];
    c2[gv] = s;
    counts[gv] = 0u;
}

// ---------------------------------------------------------------------------
// Stage 1: h = hidden @ W + b  (bf16 WMMA, f32 accumulate)
//   block = 256 threads (8 waves), tile = 64 rows x 256 cols
//   wave w: M-tile = w>>1 (16 rows), N-half = w&1 (128 cols == one group)
// B operand is double-buffered so WMMA j overlaps the loads for tile j+1;
// the next K-slab of `hidden` is prefetched while this slab's WMMAs run.
// Outputs: h (bf16, ws) and x2[token][g] (f32, ws) from f32 accumulators.
// ---------------------------------------------------------------------------
__global__ void __launch_bounds__(256)
proj_kernel(const float* __restrict__ hidden, const float* __restrict__ bvec,
            const __bf16* __restrict__ wp, __bf16* __restrict__ hbf,
            float* __restrict__ x2) {
    __shared__ __align__(16) __bf16 As[64 * 32];          // 4 KB A-stage

    const int tid  = threadIdx.x;
    const int w    = tid >> 5;
    const int lane = tid & 31;
    const int half = lane >> 4;
    const int l15  = lane & 15;
    const int mt   = w >> 1;                              // 0..3
    const int nh   = w & 1;                               // group / N-half
    const int row0 = blockIdx.x * 64;

    const int ldr  = tid >> 2;                            // LDS fill: row 0..63
    const int ldc  = (tid & 3) * 8;                       // col 0,8,16,24
    const float* hrow = hidden + (size_t)(row0 + ldr) * HID + ldc;

    v8f acc[8];
    #pragma unroll
    for (int j = 0; j < 8; ++j) acc[j] = {};

    for (int kb = 0; kb < 16; ++kb) {
        // cooperative load of hidden[row0:row0+64, kb*32 : +32] -> bf16 LDS
        const float4* hp = (const float4*)(hrow + kb * 32);
        float4 f0 = hp[0], f1 = hp[1];
        if (kb < 15)                                      // prefetch next slab
            __builtin_prefetch(hrow + (kb + 1) * 32, 0, 0);
        __bf16* d = As + ldr * 32 + ldc;
        d[0] = f2bf(f0.x); d[1] = f2bf(f0.y); d[2] = f2bf(f0.z); d[3] = f2bf(f0.w);
        d[4] = f2bf(f1.x); d[5] = f2bf(f1.y); d[6] = f2bf(f1.z); d[7] = f2bf(f1.w);
        __syncthreads();

        // A operand: 16 contiguous K per lane half (matches packed B)
        const v8bf* ap = (const v8bf*)(As + (mt * 16 + l15) * 32);
        v8bf alo = ap[2 * half];     // K = 16*half .. +7
        v8bf ahi = ap[2 * half + 1]; // K = 16*half + 8 .. +7
        v16bf a = __builtin_shufflevector(alo, ahi,
                    0,1,2,3,4,5,6,7,8,9,10,11,12,13,14,15);

        // double-buffered B operand stream
        const __bf16* wb = wp + (((size_t)kb * 16 + nh * 8) * 32 + lane) * 16;
        v16bf bcur = *(const v16bf*)(wb);
        #pragma unroll
        for (int j = 0; j < 8; ++j) {
            v16bf bnext = (j < 7) ? *(const v16bf*)(wb + (size_t)(j + 1) * 32 * 16)
                                  : bcur;
            acc[j] = __builtin_amdgcn_wmma_f32_16x16x32_bf16(
                         false, a, false, bcur, (short)0, acc[j], false, false);
            bcur = bnext;
        }
        __syncthreads();
    }

    // epilogue: bias, bf16 store, x2 partials from exact f32 accumulators
    float x2s[8];
    #pragma unroll
    for (int r = 0; r < 8; ++r) x2s[r] = 0.f;

    #pragma unroll
    for (int j = 0; j < 8; ++j) {
        int n  = (nh * 8 + j) * 16 + l15;
        float bv = bvec[n];
        #pragma unroll
        for (int r = 0; r < 8; ++r) {
            float c = acc[j][r] + bv;
            int grow = row0 + mt * 16 + 8 * half + r;     // C layout: M = 8*half+r
            hbf[(size_t)grow * DOUT + n] = f2bf(c);
            x2s[r] += c * c;
        }
    }
    #pragma unroll
    for (int r = 0; r < 8; ++r) {
        float s = x2s[r];
        s += __shfl_xor(s, 1, 32);
        s += __shfl_xor(s, 2, 32);
        s += __shfl_xor(s, 4, 32);
        s += __shfl_xor(s, 8, 32);
        if (l15 == 0)
            x2[(size_t)(row0 + mt * 16 + 8 * half + r) * NGRP + nh] = s;
    }
}

// ---------------------------------------------------------------------------
// Stage 2: distances + argmin + encodings + quantized gather + usage counts.
//   block = 128 threads (4 waves); wave = one 16-token tile for one group.
// All 4 K-step B tiles of a V-tile are loaded as one clause before the
// 4-WMMA accumulation chain.
// ---------------------------------------------------------------------------
__global__ void __launch_bounds__(128)
dist_kernel(const __bf16* __restrict__ hbf, const __bf16* __restrict__ cp,
            const float* __restrict__ x2, const float* __restrict__ c2v,
            const float* __restrict__ codevectors, float* __restrict__ out,
            unsigned* __restrict__ counts) {
    const int tid  = threadIdx.x;
    const int w    = tid >> 5;
    const int lane = tid & 31;
    const int half = lane >> 4;
    const int l15  = lane & 15;
    const int g    = blockIdx.y;
    const int t0   = (blockIdx.x * 4 + w) * 16;

    // A operand (h tile 16x128 bf16), all 4 K-steps up front
    v16bf a[KS2];
    const __bf16* arow = hbf + (size_t)(t0 + l15) * DOUT + g * GD;
    #pragma unroll
    for (int ks = 0; ks < KS2; ++ks) {
        const __bf16* p = arow + ks * 32;
        v8bf lo = *(const v8bf*)(p + 16 * half);
        v8bf hi = *(const v8bf*)(p + 16 * half + 8);
        a[ks] = __builtin_shufflevector(lo, hi,
                    0,1,2,3,4,5,6,7,8,9,10,11,12,13,14,15);
    }

    float x2r[8];
    #pragma unroll
    for (int r = 0; r < 8; ++r)
        x2r[r] = x2[(size_t)(t0 + 8 * half + r) * NGRP + g];

    float runv[8]; int runi[8];
    #pragma unroll
    for (int r = 0; r < 8; ++r) { runv[r] = 3.4e38f; runi[r] = 0; }

    float* dist_out = out + DIST_BASE;
    const __bf16* cpg = cp + (((size_t)g * VT * KS2) * 32 + lane) * 16;

    for (int vt = 0; vt < VT; ++vt) {
        float c2l = c2v[g * NCV + vt * 16 + l15];

        v16bf b[KS2];                                     // one load clause
        #pragma unroll
        for (int ks = 0; ks < KS2; ++ks)
            b[ks] = *(const v16bf*)(cpg + ((size_t)vt * KS2 + ks) * 32 * 16);

        v8f acc = {};
        #pragma unroll
        for (int ks = 0; ks < KS2; ++ks)
            acc = __builtin_amdgcn_wmma_f32_16x16x32_bf16(
                      false, a[ks], false, b[ks], (short)0, acc, false, false);

        int v = vt * 16 + l15;
        #pragma unroll
        for (int r = 0; r < 8; ++r) {
            float dv = x2r[r] + c2l - 2.0f * acc[r];
            int row = t0 + 8 * half + r;
            dist_out[((size_t)g * NTOK + row) * NCV + v] = dv;
            if (dv < runv[r]) { runv[r] = dv; runi[r] = v; }   // first-min kept
        }
    }

    // cross-lane argmin within each 16-lane half (ties -> smaller index)
    int argi[8];
    #pragma unroll
    for (int r = 0; r < 8; ++r) {
        float bv = runv[r]; int bi = runi[r];
        #pragma unroll
        for (int m = 1; m <= 8; m <<= 1) {
            float ov = __shfl_xor(bv, m, 32);
            int   oi = __shfl_xor(bi, m, 32);
            if (ov < bv || (ov == bv && oi < bi)) { bv = ov; bi = oi; }
        }
        argi[r] = bi;
    }

    // one-hot encodings
    float* enc_out = out + ENC_BASE;
    #pragma unroll
    for (int r = 0; r < 8; ++r) {
        int row = t0 + 8 * half + r;
        float* e = enc_out + ((size_t)g * NTOK + row) * NCV;
        #pragma unroll
        for (int vt = 0; vt < VT; ++vt) {
            int v = vt * 16 + l15;
            e[v] = (v == argi[r]) ? 1.0f : 0.0f;
        }
    }

    // quantized features: gather winning codevector (f32)
    #pragma unroll
    for (int r = 0; r < 8; ++r) {
        int row = t0 + 8 * half + r;
        const float* cvrow = codevectors + ((size_t)g * NCV + argi[r]) * GD;
        float* q = out + QF_BASE + (size_t)row * DOUT + g * GD;
        #pragma unroll
        for (int j = 0; j < 8; ++j) {
            int dd = j * 16 + l15;
            q[dd] = cvrow[dd];
        }
    }

    // usage histogram for perplexity
    if (l15 == 0) {
        #pragma unroll
        for (int r = 0; r < 8; ++r)
            atomicAdd(&counts[g * NCV + argi[r]], 1u);
    }
}

// ---------------------------------------------------------------------------
// Stage 3: perplexity (single wave).
// ---------------------------------------------------------------------------
__global__ void perp_kernel(const unsigned* __restrict__ counts, float* __restrict__ out) {
    int lane = threadIdx.x;                                // 32 threads
    const float inv = 1.0f / (float)NTOK;
    float psum = 0.f;
    for (int g = 0; g < NGRP; ++g) {
        float s = 0.f;
        for (int v = lane; v < NCV; v += 32) {
            float p = (float)counts[g * NCV + v] * inv;
            p = fminf(fmaxf(p, 1e-10f), 1.0f);
            s += p * logf(p + 1e-10f);
        }
        s += __shfl_xor(s, 1, 32);
        s += __shfl_xor(s, 2, 32);
        s += __shfl_xor(s, 4, 32);
        s += __shfl_xor(s, 8, 32);
        s += __shfl_xor(s, 16, 32);
        psum += expf(-s);
    }
    if (lane == 0) out[PERP_IDX] = psum * (1.0f / NGRP);
}

// ---------------------------------------------------------------------------
extern "C" void kernel_launch(void* const* d_in, const int* in_sizes, int n_in,
                              void* d_out, int out_size, void* d_ws, size_t ws_size,
                              hipStream_t stream) {
    (void)in_sizes; (void)n_in; (void)out_size; (void)ws_size;

    const float* hidden = (const float*)d_in[0];
    const float* W      = (const float*)d_in[1];
    const float* bvec   = (const float*)d_in[2];
    const float* cv     = (const float*)d_in[3];
    float* out          = (float*)d_out;

    char* ws = (char*)d_ws;
    __bf16*   hbf    = (__bf16*)(ws + WS_HBF);
    float*    x2     = (float*)(ws + WS_X2);
    __bf16*   wpack  = (__bf16*)(ws + WS_WPACK);
    __bf16*   cpack  = (__bf16*)(ws + WS_CPACK);
    float*    c2     = (float*)(ws + WS_C2);
    unsigned* counts = (unsigned*)(ws + WS_CNT);

    prepack_w_kernel<<<32, 256, 0, stream>>>(W, wpack);
    prepack_c_kernel<<<20, 256, 0, stream>>>(cv, cpack);
    c2_counts_kernel<<<3, 256, 0, stream>>>(cv, c2, counts);

    proj_kernel<<<NTOK / 64, 256, 0, stream>>>(hidden, bvec, wpack, hbf, x2);

    dist_kernel<<<dim3(NTOK / 64, NGRP), 128, 0, stream>>>(
        hbf, cpack, x2, c2, cv, out, counts);

    perp_kernel<<<1, 32, 0, stream>>>(counts, out);
}
